// SimCLR_tau_71528385347973
// MI455X (gfx1250) — compile-verified, matrix operand order checked by reading, not run on hardware
//
#include <hip/hip_runtime.h>
#include <math.h>

typedef _Float16 half4  __attribute__((ext_vector_type(4)));
typedef _Float16 half8  __attribute__((ext_vector_type(8)));
typedef _Float16 half16 __attribute__((ext_vector_type(16)));
typedef float    float4v __attribute__((ext_vector_type(4)));
typedef float    float8v __attribute__((ext_vector_type(8)));

#define B_ROWS 8192
#define N_ROWS 16384   // 2B
#define DIM    128
#define NTILES (N_ROWS / 16)   // 1024 column tiles
#define NQUART 4               // column range split; 4096 waves total
#define QTILES (NTILES / NQUART)

// scale = sqrt( (1/TEMP) * log2(e) ) so that dot(zn_i, zn_j) = sim/TEMP*log2e
// -> Sum exp(sim/TEMP) == Sum exp2(dot), and pos = dot * ln2.
#define NORM_SCALE 1.6986436f   // sqrt(2 * 1.4426950408889634)
#define LN2F       0.6931471805599453f

// raw v_exp_f32 (native exp2)
__device__ __forceinline__ float fast_exp2(float x) {
  return __builtin_amdgcn_exp2f(x);
}

// ---------------------------------------------------------------------------
// Kernel 1: row-normalize concat(z1,z2), fold in NORM_SCALE, store f16.
// One wave per row. Also zeroes per-row accumulators and the scalar output.
// ---------------------------------------------------------------------------
__global__ __launch_bounds__(256)
void normalize_f16_kernel(const float* __restrict__ z1,
                          const float* __restrict__ z2,
                          _Float16* __restrict__ zn,
                          float* __restrict__ rowSum,
                          float* __restrict__ rowPos,
                          float* __restrict__ out) {
  if (blockIdx.x == 0 && threadIdx.x == 0) out[0] = 0.0f;

  const int row  = (blockIdx.x * blockDim.x + threadIdx.x) >> 5;
  const int lane = threadIdx.x & 31;
  if (row >= N_ROWS) return;
  if (lane == 0) { rowSum[row] = 0.0f; rowPos[row] = 0.0f; }

  const float* __restrict__ src =
      (row < B_ROWS) ? (z1 + (size_t)row * DIM)
                     : (z2 + (size_t)(row - B_ROWS) * DIM);

  float4v v = *(const float4v*)(src + lane * 4);
  float ss = v.x * v.x + v.y * v.y + v.z * v.z + v.w * v.w;
#pragma unroll
  for (int m = 1; m < 32; m <<= 1) ss += __shfl_xor(ss, m, 32);

  const float scale = NORM_SCALE * rsqrtf(ss);

  half4 h;
  h.x = (_Float16)(v.x * scale);
  h.y = (_Float16)(v.y * scale);
  h.z = (_Float16)(v.z * scale);
  h.w = (_Float16)(v.w * scale);
  *(half4*)(zn + (size_t)row * DIM + lane * 4) = h;
}

// ---------------------------------------------------------------------------
// Fragment helpers.  16-bit A/B layout (ISA 7.12.2): lane L holds row/col
// L%16, K-halves { kk*32 + hi*8 .. +7 , kk*32 + 16 + hi*8 .. +7 }, hi=L/16.
// ---------------------------------------------------------------------------
__device__ __forceinline__ void loadFrag4(half16 (&f)[4],
                                          const _Float16* __restrict__ zn,
                                          int rowIdx, int hi) {
  const _Float16* base = zn + (size_t)rowIdx * DIM + hi * 8;
#pragma unroll
  for (int kk = 0; kk < 4; ++kk) {
    const half8* p = (const half8*)(base + kk * 32);
    half8 lo  = p[0];
    half8 hi8 = p[2];
    f[kk] = __builtin_shufflevector(lo, hi8,
              0, 1, 2, 3, 4, 5, 6, 7, 8, 9, 10, 11, 12, 13, 14, 15);
  }
}

__device__ __forceinline__ float8v wmmaTile(const half16 (&a)[4],
                                            const half16 (&b)[4]) {
  float8v acc = {};
#pragma unroll
  for (int kk = 0; kk < 4; ++kk) {
    acc = __builtin_amdgcn_wmma_f32_16x16x32_f16(
        false, a[kk], false, b[kk], (short)0, acc, false, false);
  }
  return acc;
}

// C/D f32 layout: VGPR m -> row (m + 8*hi), column N = lane%16.
// `special` is wave-uniform (scalar) -> s_cbranch, no exec-mask games.
__device__ __forceinline__ void processTile(const float8v& acc, int ct,
                                            int rowBase, int hi, int lm,
                                            bool special,
                                            float (&accExp)[8],
                                            float (&posAcc)[8]) {
  if (__builtin_expect(!special, 1)) {
#pragma unroll
    for (int m = 0; m < 8; ++m) accExp[m] += fast_exp2(acc[m]);
  } else {
    const int gCol = ct * 16 + lm;
#pragma unroll
    for (int m = 0; m < 8; ++m) {
      const int gRow = rowBase + m + hi * 8;
      const float v = acc[m];                   // sim/TEMP * log2e
      float e = fast_exp2(v);
      if (gRow == gCol) e = 0.0f;               // diagonal mask
      accExp[m] += e;
      if (gCol == ((gRow + B_ROWS) & (N_ROWS - 1))) posAcc[m] += v;
    }
  }
}

// ---------------------------------------------------------------------------
// Kernel 2: partial NT-Xent. 4096 waves = 1024 row-strips x 4 column
// quarters. Double-buffered B fragments + one-tile-deep epilogue pipelining:
// tile ct's exp work overlaps tile ct+1's WMMA chain (hides the
// WMMA->VALU RAW hazard) while tile ct+2's loads are in flight.
// ---------------------------------------------------------------------------
__global__ __launch_bounds__(256)
void nt_xent_wmma_kernel(const _Float16* __restrict__ zn,
                         float* __restrict__ rowSum,
                         float* __restrict__ rowPos) {
  const int lane = threadIdx.x & 31;
  // wave index is uniform across the wave: force into SGPRs so tile-special
  // tests and loop control are scalar.
  const int gw = __builtin_amdgcn_readfirstlane(
      (int)(blockIdx.x * 8 + (threadIdx.x >> 5)));   // 512 blocks * 8 waves
  const int strip = gw & (NTILES - 1);               // consecutive in block
  const int q     = gw >> 10;                        // same quarter in block
  const int rowBase = strip * 16;
  const int hi = lane >> 4;
  const int lm = lane & 15;

  const int ctBase = q * QTILES;
  const int ctEnd  = ctBase + QTILES;
  const int diagTile = strip;
  const int posTile  = (strip + NTILES / 2) & (NTILES - 1);

  half16 aFrag[4];
  loadFrag4(aFrag, zn, rowBase + lm, hi);            // A strip in registers

  float accExp[8], posAcc[8];
#pragma unroll
  for (int m = 0; m < 8; ++m) { accExp[m] = 0.0f; posAcc[m] = 0.0f; }

  half16 b0[4], b1[4];
  loadFrag4(b0, zn, ctBase * 16 + lm, hi);           // frags tile ctBase
  loadFrag4(b1, zn, (ctBase + 1) * 16 + lm, hi);     // frags tile ctBase+1
  float8v accA = wmmaTile(aFrag, b0);                // acc for tile ctBase

  // invariant at loop top: accA = tile ct, b1 = frags for tile ct+1
  for (int ct = ctBase; ct < ctEnd - 2; ct += 2) {
    loadFrag4(b0, zn, (ct + 2) * 16 + lm, hi);       // prefetch ct+2
    float8v accB = wmmaTile(aFrag, b1);              // compute ct+1
    processTile(accA, ct, rowBase, hi, lm,
                (ct == diagTile) || (ct == posTile), accExp, posAcc);

    loadFrag4(b1, zn, (ct + 3) * 16 + lm, hi);       // prefetch ct+3
    accA = wmmaTile(aFrag, b0);                      // compute ct+2
    processTile(accB, ct + 1, rowBase, hi, lm,
                (ct + 1 == diagTile) || (ct + 1 == posTile), accExp, posAcc);
  }
  // tail: accA = tile ctEnd-2, b1 = frags tile ctEnd-1
  {
    const int ct = ctEnd - 2;
    float8v accB = wmmaTile(aFrag, b1);
    processTile(accA, ct, rowBase, hi, lm,
                (ct == diagTile) || (ct == posTile), accExp, posAcc);
    processTile(accB, ct + 1, rowBase, hi, lm,
                (ct + 1 == diagTile) || (ct + 1 == posTile), accExp, posAcc);
  }

  // reduce across the 16 column-lanes of each half-wave; lane lm==m keeps row m
  float sRow = 0.0f, pRow = 0.0f;
#pragma unroll
  for (int m = 0; m < 8; ++m) {
    float s = accExp[m];
    float p = posAcc[m];
#pragma unroll
    for (int msk = 1; msk < 16; msk <<= 1) {
      s += __shfl_xor(s, msk, 32);
      p += __shfl_xor(p, msk, 32);
    }
    if (lm == m) { sRow = s; pRow = p; }
  }
  if (lm < 8) {
    const int gRow = rowBase + lm + hi * 8;
    atomicAdd(&rowSum[gRow], sRow);
    atomicAdd(&rowPos[gRow], pRow * LN2F);   // back to natural-log units
  }
}

// ---------------------------------------------------------------------------
// Kernel 3: loss_i = log(sumExp_i) - pos_i ; mean over rows into scalar.
// ---------------------------------------------------------------------------
__global__ __launch_bounds__(256)
void finalize_kernel(const float* __restrict__ rowSum,
                     const float* __restrict__ rowPos,
                     float* __restrict__ out) {
  const int i = blockIdx.x * blockDim.x + threadIdx.x;   // 16384 threads
  const int lane = threadIdx.x & 31;
  float loss = __logf(rowSum[i]) - rowPos[i];
#pragma unroll
  for (int m = 1; m < 32; m <<= 1) loss += __shfl_xor(loss, m, 32);
  if (lane == 0) atomicAdd(out, loss * (1.0f / (float)N_ROWS));
}

// ---------------------------------------------------------------------------
extern "C" void kernel_launch(void* const* d_in, const int* in_sizes, int n_in,
                              void* d_out, int out_size, void* d_ws, size_t ws_size,
                              hipStream_t stream) {
  (void)in_sizes; (void)n_in; (void)out_size; (void)ws_size;
  const float* z1 = (const float*)d_in[0];
  const float* z2 = (const float*)d_in[1];
  float* out = (float*)d_out;

  _Float16* zn   = (_Float16*)d_ws;                              // 4 MB
  float* rowSum  = (float*)((char*)d_ws + (size_t)N_ROWS * DIM * 2);
  float* rowPos  = rowSum + N_ROWS;                              // +64 KB

  normalize_f16_kernel<<<N_ROWS / 8, 256, 0, stream>>>(z1, z2, zn,
                                                       rowSum, rowPos, out);
  nt_xent_wmma_kernel<<<512, 256, 0, stream>>>(zn, rowSum, rowPos);
  finalize_kernel<<<N_ROWS / 256, 256, 0, stream>>>(rowSum, rowPos, out);
}
